// ConvTemporalGraphical_45938970198098
// MI455X (gfx1250) — compile-verified, matrix operand order checked by reading, not run on hardware
//
#include <hip/hip_runtime.h>
#include <hip/hip_bf16.h>

#define NN   4
#define CIN  64
#define CO   64
#define TT   128
#define VV   24
#define KK   3
#define BD   16
#define TKW  9
#define TP   4

typedef __attribute__((ext_vector_type(16))) _Float16 v16h;
typedef __attribute__((ext_vector_type(8)))  float    v8f;
typedef _Float16 half_t;

// ---- WMMA fragment helpers (wave32, v_wmma_f32_16x16x32_f16) ----
// A-matrix 16x32 layout: lanes 0-15: M=lane, elems 0..7 = K 0..7, elems 8..15 = K 16..23
//                        lanes 16-31: M=lane-16, K offsets +8.
__device__ __forceinline__ v16h load_a_lds(const half_t* base, int mt, int kk, int lane, int ld) {
  int row = mt * 16 + (lane & 15);
  int kb  = kk * 32 + ((lane & 16) ? 8 : 0);
  const half_t* p = base + row * ld + kb;
  v16h a;
#pragma unroll
  for (int i = 0; i < 8; ++i) { a[i] = p[i]; a[8 + i] = p[16 + i]; }
  return a;
}

// B-matrix 32x16 (KxN) from f32 source (row-major K x ld), converted to f16.
__device__ __forceinline__ v16h load_b_f32(const float* W, int nt, int kk, int lane, int ld) {
  int col = nt * 16 + (lane & 15);
  int kb  = kk * 32 + ((lane & 16) ? 8 : 0);
  v16h b;
#pragma unroll
  for (int i = 0; i < 8; ++i) {
    b[i]     = (half_t)W[(kb + i)      * ld + col];
    b[8 + i] = (half_t)W[(kb + 16 + i) * ld + col];
  }
  return b;
}

// B fragment for W1[64:80] (K=16 real rows, rows 16..31 zero-padded).
__device__ __forceinline__ v16h load_b_w1b(const float* W1, int nt, int lane) {
  int col = nt * 16 + (lane & 15);
  int kb  = (lane & 16) ? 8 : 0;
  v16h b;
#pragma unroll
  for (int i = 0; i < 8; ++i) {
    b[i]     = (half_t)W1[(CO + kb + i) * CO + col];  // K = kb+i  (< 16)
    b[8 + i] = (half_t)0.0f;                          // K = 16+kb+i (zero pad)
  }
  return b;
}

// D/C 16x16 f32: lanes 0-15: N=lane, vgpr r -> M=r; lanes 16-31: N=lane-16, M=r+8.
__device__ __forceinline__ void store_d_tile(float* dst, int rows, v8f d, int mt, int nt,
                                             int lane, const float* __restrict__ bias) {
  int col   = nt * 16 + (lane & 15);
  int rbase = mt * 16 + ((lane & 16) ? 8 : 0);
  float bv = bias ? bias[col] : 0.0f;
#pragma unroll
  for (int r = 0; r < 8; ++r) {
    int row = rbase + r;
    if (row < rows) dst[row * CO + col] = d[r] + bv;
  }
}

// ---------------- Kernel 1: temporal conv -> y (f16, per-(n,t,k) 24x64 row-major tiles)
__global__ void __launch_bounds__(256) conv_kernel(const float* __restrict__ x,
                                                   const float* __restrict__ cw,
                                                   const float* __restrict__ cb,
                                                   half_t* __restrict__ yws) {
  __shared__ float xs[CIN * VV];
  int tid = threadIdx.x;
  int bx  = blockIdx.x;
  int n = bx / TT, t = bx % TT;

  float acc[18];
#pragma unroll
  for (int j = 0; j < 18; ++j) acc[j] = 0.0f;

#pragma unroll 1
  for (int dt = 0; dt < TKW; ++dt) {
    int tt = t + dt - TP;
    __syncthreads();
    bool inb = (tt >= 0) && (tt < TT);
    for (int i = tid; i < CIN * VV; i += 256) {
      int ci = i / VV, w = i % VV;
      xs[i] = inb ? x[(((size_t)n * CIN + ci) * TT + tt) * VV + w] : 0.0f;
    }
    // prefetch next temporal slice (global_prefetch_b8)
    int tn = t + dt + 1 - TP;
    if ((dt + 1 < TKW) && tn >= 0 && tn < TT)
      __builtin_prefetch(&x[(((size_t)n * CIN + (tid & 63)) * TT + tn) * VV], 0, 3);
    __syncthreads();
#pragma unroll 1
    for (int j = 0; j < 18; ++j) {
      int o = j * 256 + tid;          // 4608 outputs = (k*64+c) x w
      int kc = o / VV, w = o % VV;
      const float* cwp = cw + (size_t)kc * CIN * TKW + dt;
      float s = acc[j];
#pragma unroll 4
      for (int ci = 0; ci < CIN; ++ci) s += xs[ci * VV + w] * cwp[ci * TKW];
      acc[j] = s;
    }
  }

  size_t base = (size_t)(n * TT + t) * (KK * VV * CO);
#pragma unroll 1
  for (int j = 0; j < 18; ++j) {
    int o = j * 256 + tid;
    int kc = o / VV, w = o % VV;
    int k = kc / CO, c = kc % CO;
    yws[base + ((size_t)k * VV + w) * CO + c] = (half_t)(acc[j] + cb[kc]);
  }
}

// ---------------- Kernel 2: fused edge-MLP + spatial aggregation, one block per (n,t)
__global__ void __launch_bounds__(128) fused_kernel(const half_t* __restrict__ yws,
                                                    const float* __restrict__ bin,
                                                    const float* __restrict__ W1,
                                                    const float* __restrict__ b1,
                                                    const float* __restrict__ W2,
                                                    const float* __restrict__ b2,
                                                    const float* __restrict__ Amat,
                                                    float* __restrict__ out) {
  __shared__ half_t ybuf[KK][32][CO];    // y tiles, rows 24..31 = 0     (12 KB)
  __shared__ half_t binh[32][32];        // bin_v f16, padded             (2 KB)
  __shared__ float  xwbuf[KK][VV][CO];   // xw + b1, per k               (18 KB)
  __shared__ float  bwv[VV][CO];         // bw for current v              (6 KB)
  __shared__ half_t h16[32][CO];         // h tile, rows 24..31 = 0       (4 KB)
  __shared__ float  resbuf[VV][CO];      // res tile                      (6 KB)

  int tid  = threadIdx.x;
  int lane = tid & 31, wave = tid >> 5;  // 4 wave32s; wave = N-tile index
  int bx = blockIdx.x;
  int n = bx / TT, t = bx % TT;

  // ---- async-copy the (n,t) y tiles into LDS (ASYNCcnt path) ----
  {
    const half_t* ysrc = yws + (size_t)(n * TT + t) * (KK * VV * CO);
    unsigned long long gb = (unsigned long long)(uintptr_t)ysrc;
    unsigned ldsY = (unsigned)(uintptr_t)(void*)&ybuf[0][0][0];
    // 3*24*64 halfs = 9216 B = 576 x b128 chunks
    for (int j = tid; j < (KK * VV * CO * 2) / 16; j += 128) {
      int k    = j / 192;               // 192 chunks per k-slab
      int rem  = j - k * 192;
      int w    = rem >> 3;
      int cby  = (rem & 7) * 16;        // byte offset within 128-B row
      unsigned lo = ldsY + (unsigned)((k * 32 + w) * (CO * 2) + cby);
      unsigned go = (unsigned)(j * 16);
      asm volatile("global_load_async_to_lds_b128 %0, %1, %2"
                   :: "v"(lo), "v"(go), "s"(gb) : "memory");
    }
  }
  // zero pads while the async copy is in flight
  for (int i = tid; i < KK * 8 * CO; i += 128) {        // ybuf rows 24..31
    int k = i / (8 * CO);
    int r = (i / CO) & 7;
    int c = i % CO;
    ybuf[k][VV + r][c] = (half_t)0.0f;
  }
  for (int i = tid; i < 32 * 32; i += 128) binh[i >> 5][i & 31] = (half_t)0.0f;
  for (int i = tid; i < 8 * CO; i += 128) h16[VV + i / CO][i % CO] = (half_t)0.0f;

  // B fragments held in registers for the whole kernel (wave owns N-tile = wave)
  v16h w1f[2], w2f[2], w1bf;
#pragma unroll
  for (int kk = 0; kk < 2; ++kk) {
    w1f[kk] = load_b_f32(W1, wave, kk, lane, CO);   // W1[:64]
    w2f[kk] = load_b_f32(W2, wave, kk, lane, CO);
  }
  w1bf = load_b_w1b(W1, wave, lane);                // W1[64:80], K-padded

  asm volatile("s_wait_asynccnt 0x0" ::: "memory");
  __syncthreads();

  // ---- xw[k] = Y_k(24x64) @ W1a(64x64) + b1, via WMMA from LDS ----
#pragma unroll 1
  for (int k = 0; k < KK; ++k) {
#pragma unroll
    for (int mt = 0; mt < 2; ++mt) {
      v8f acc = {};
#pragma unroll
      for (int kk = 0; kk < 2; ++kk) {
        v16h a = load_a_lds(&ybuf[k][0][0], mt, kk, lane, CO);
        acc = __builtin_amdgcn_wmma_f32_16x16x32_f16(false, a, false, w1f[kk],
                                                     (short)0, acc, false, false);
      }
      store_d_tile(&xwbuf[k][0][0], VV, acc, mt, wave, lane, b1);
    }
  }
  __syncthreads();

  const float* binb = bin + (size_t)(n * TT + t) * (VV * VV * BD);

#pragma unroll 1
  for (int v = 0; v < VV; ++v) {
    // stage bin_v (24x16) as f16 into padded 32x32 LDS tile
    for (int i = tid; i < VV * BD; i += 128) {
      int w = i >> 4, b = i & 15;
      binh[w][b] = (half_t)binb[(size_t)(v * VV + w) * BD + b];
    }
    __syncthreads();

    // bw_v = bin_v(24x16,pad32) @ W1b(16x64,pad32) via WMMA
#pragma unroll
    for (int mt = 0; mt < 2; ++mt) {
      v8f acc = {};
      v16h a = load_a_lds(&binh[0][0], mt, 0, lane, 32);
      acc = __builtin_amdgcn_wmma_f32_16x16x32_f16(false, a, false, w1bf,
                                                   (short)0, acc, false, false);
      store_d_tile(&bwv[0][0], VV, acc, mt, wave, lane, nullptr);
    }
    __syncthreads();

    float osum = 0.0f;   // per-thread output accumulator (c = tid for tid<64)
#pragma unroll 1
    for (int k = 0; k < KK; ++k) {
      // h = relu(xw_k + bw_v) -> f16 LDS in WMMA-A layout
      for (int i = tid; i < VV * CO; i += 128) {
        int w = i >> 6, d = i & 63;
        float hv = xwbuf[k][w][d] + bwv[w][d];
        h16[w][d] = (half_t)(hv > 0.0f ? hv : 0.0f);
      }
      __syncthreads();

      // res = h(24x64) @ W2(64x64) + b2, via WMMA
#pragma unroll
      for (int mt = 0; mt < 2; ++mt) {
        v8f acc = {};
#pragma unroll
        for (int kk = 0; kk < 2; ++kk) {
          v16h a = load_a_lds(&h16[0][0], mt, kk, lane, CO);
          acc = __builtin_amdgcn_wmma_f32_16x16x32_f16(false, a, false, w2f[kk],
                                                       (short)0, acc, false, false);
        }
        store_d_tile(&resbuf[0][0], VV, acc, mt, wave, lane, b2);
      }
      __syncthreads();

      // out[n,c,t,v] += sum_w res[w][c] * A[k,v,w]
      if (tid < CO) {
        const float* Ak = Amat + ((size_t)k * VV + v) * VV;
        float s = 0.0f;
#pragma unroll
        for (int w = 0; w < VV; ++w) s += resbuf[w][tid] * Ak[w];
        osum += s;
      }
      __syncthreads();
    }

    if (tid < CO)
      out[(((size_t)n * CO + tid) * TT + t) * VV + v] = osum;
    __syncthreads();
  }
}

// ---------------- Kernel 3: tuple's second output = A passthrough
__global__ void copy_kernel(const float* __restrict__ src, float* __restrict__ dst, int n) {
  int i = blockIdx.x * blockDim.x + threadIdx.x;
  if (i < n) dst[i] = src[i];
}

extern "C" void kernel_launch(void* const* d_in, const int* in_sizes, int n_in,
                              void* d_out, int out_size, void* d_ws, size_t ws_size,
                              hipStream_t stream) {
  const float* x   = (const float*)d_in[0];
  const float* A   = (const float*)d_in[1];
  const float* bin = (const float*)d_in[2];
  const float* cw  = (const float*)d_in[3];
  const float* cb  = (const float*)d_in[4];
  const float* W1  = (const float*)d_in[5];
  const float* b1  = (const float*)d_in[6];
  const float* W2  = (const float*)d_in[7];
  const float* b2  = (const float*)d_in[8];
  float* out = (float*)d_out;

  half_t* yws = (half_t*)d_ws;   // N*T*K*V*CO f16 = ~4.7 MB

  conv_kernel<<<NN * TT, 256, 0, stream>>>(x, cw, cb, yws);
  fused_kernel<<<NN * TT, 128, 0, stream>>>(yws, bin, W1, b1, W2, b2, A, out);

  const int outElems = NN * CO * TT * VV;   // 786432
  const int aElems   = KK * VV * VV;        // 1728
  copy_kernel<<<(aElems + 255) / 256, 256, 0, stream>>>(A, out + outElems, aElems);
}
